// Attention_52871047414112
// MI455X (gfx1250) — matrix operand order repaired and checked by reading, not compiled, not run on hardware
//
#include <hip/hip_runtime.h>
#include <hip/hip_bf16.h>

// ---------------------------------------------------------------------------
// Problem constants (match reference)
// ---------------------------------------------------------------------------
#define DIMC   1024
#define NHEAD  16
#define SEQ    2048
#define NBATCH 4
#define HDIM   64            // DIMC / NHEAD
#define ATTN_SCALE 0.125f    // HDIM^-0.5

typedef __attribute__((ext_vector_type(16))) __bf16 v16bf;
typedef __attribute__((ext_vector_type(8)))  float  v8f;

// Fragment viewed as bf16x16 (8 VGPRs), or raw words for memory staging.
union Frag {
    v16bf          v;
    unsigned short us[16];
    unsigned int   ui[8];
    uint4          q4[2];
};

// float -> bf16 via native conversion (backend selects hw cvt on gfx1250)
__device__ __forceinline__ __bf16 tobf(float f) { return (__bf16)f; }

// Row reductions across a 16-lane group (wave32; xor masks < 16 stay in-half)
__device__ __forceinline__ float rowmax16(float x) {
#pragma unroll
    for (int m = 1; m < 16; m <<= 1) x = fmaxf(x, __shfl_xor(x, m, 32));
    return x;
}
__device__ __forceinline__ float rowsum16(float x) {
#pragma unroll
    for (int m = 1; m < 16; m <<= 1) x += __shfl_xor(x, m, 32);
    return x;
}

// ---------------------------------------------------------------------------
// Kernel 0: one-shot fp32 -> bf16 conversion (bandwidth bound, done ONCE,
// so the GEMM hot loop never converts).
// ---------------------------------------------------------------------------
__global__ __launch_bounds__(256)
void cvt_bf16_kernel(const float* __restrict__ src, __bf16* __restrict__ dst, int n4)
{
    const int i = blockIdx.x * blockDim.x + threadIdx.x;
    if (i >= n4) return;
    const float4 f = ((const float4*)src)[i];
    union { __bf16 b[4]; uint2 u; } r;
    r.b[0] = tobf(f.x); r.b[1] = tobf(f.y); r.b[2] = tobf(f.z); r.b[3] = tobf(f.w);
    ((uint2*)dst)[i] = r.u;
}

// ---------------------------------------------------------------------------
// Kernel 1: QKV projection GEMM, bf16 inputs, fp32 accumulate.
//   out[m][n] = sum_k x[m][k] * w[n][k] + b[n]    (M=8192, N=3072, K=1024)
// Each wave computes a 32x32 tile (2x2 WMMA accumulators): per k-step
// 2 A-frags + 2 B-frags (uint4 loads, no conversion) -> 4 WMMAs.
// Results scattered to K:[B,H,T,D], Q:[B,H,T,D] (pre-scaled), V:[B,H,D,T].
// ---------------------------------------------------------------------------
__global__ __launch_bounds__(128)
void qkv_gemm_kernel(const __bf16* __restrict__ xb, const __bf16* __restrict__ wb,
                     const float* __restrict__ bias,
                     __bf16* __restrict__ kbuf, __bf16* __restrict__ qbuf,
                     __bf16* __restrict__ vbuf)
{
    const int lane = threadIdx.x & 31;
    const int wave = threadIdx.x >> 5;
    const int half = lane >> 4;      // 0 or 1 (lane group)
    const int l16  = lane & 15;

    const int NTN  = (3 * DIMC) / 32;             // 96 tiles along N
    const int tile = blockIdx.x * 4 + wave;       // 256*96 = 24576 tiles total
    const int tm   = tile / NTN;
    const int tn   = tile % NTN;
    const int m0   = tm * 32;
    const int n0   = tn * 32;

    const __bf16* arow0 = xb + (size_t)(m0 + l16) * DIMC;       // A rows (M)
    const __bf16* arow1 = arow0 + (size_t)16 * DIMC;
    const __bf16* brow0 = wb + (size_t)(n0 + l16) * DIMC;       // B rows (N), w is [n][k]
    const __bf16* brow1 = brow0 + (size_t)16 * DIMC;

    v8f acc00 = {}, acc01 = {}, acc10 = {}, acc11 = {};

#pragma unroll 2
    for (int k0 = 0; k0 < DIMC; k0 += 32) {
        Frag a0, a1, b0, b1;
        // A-frag (16x32): elems 0..7 <- K=k0+8*half+j ; 8..15 <- K=k0+16+8*half+j
        a0.q4[0] = *(const uint4*)(const void*)(arow0 + k0 + 8 * half);
        a0.q4[1] = *(const uint4*)(const void*)(arow0 + k0 + 16 + 8 * half);
        a1.q4[0] = *(const uint4*)(const void*)(arow1 + k0 + 8 * half);
        a1.q4[1] = *(const uint4*)(const void*)(arow1 + k0 + 16 + 8 * half);
        // B-frag (32x16): elems 0..7 <- K=k0+8*half+j ; 8..15 <- K=k0+16+8*half+j
        b0.q4[0] = *(const uint4*)(const void*)(brow0 + k0 + 8 * half);
        b0.q4[1] = *(const uint4*)(const void*)(brow0 + k0 + 16 + 8 * half);
        b1.q4[0] = *(const uint4*)(const void*)(brow1 + k0 + 8 * half);
        b1.q4[1] = *(const uint4*)(const void*)(brow1 + k0 + 16 + 8 * half);

        acc00 = __builtin_amdgcn_wmma_f32_16x16x32_bf16(false, a0.v, false, b0.v, (short)0, acc00, false, false);
        acc01 = __builtin_amdgcn_wmma_f32_16x16x32_bf16(false, a0.v, false, b1.v, (short)0, acc01, false, false);
        acc10 = __builtin_amdgcn_wmma_f32_16x16x32_bf16(false, a1.v, false, b0.v, (short)0, acc10, false, false);
        acc11 = __builtin_amdgcn_wmma_f32_16x16x32_bf16(false, a1.v, false, b1.v, (short)0, acc11, false, false);
    }

    // Epilogue: bias, q-scaling, scatter to head-major bf16 buffers.
    v8f accs[2][2] = {{acc00, acc01}, {acc10, acc11}};
#pragma unroll
    for (int nb = 0; nb < 2; ++nb) {
        const int   n     = n0 + nb * 16 + l16;
        const float bn    = bias[n];
        const int   chunk = n >> 10;      // 0=k, 1=q, 2=v
        const int   cc    = n & 1023;
        const int   h     = cc >> 6;
        const int   d     = cc & 63;
        const float mul   = (chunk == 1) ? ATTN_SCALE : 1.0f;
#pragma unroll
        for (int ma = 0; ma < 2; ++ma) {
#pragma unroll
            for (int i = 0; i < 8; ++i) {
                const int m  = m0 + ma * 16 + i + 8 * half;   // global row
                const int bi = m >> 11;                        // / SEQ
                const int t  = m & (SEQ - 1);
                const __bf16 bv = tobf((accs[ma][nb][i] + bn) * mul);
                if (chunk == 0)
                    kbuf[(((size_t)(bi * NHEAD + h)) * SEQ + t) * HDIM + d] = bv;
                else if (chunk == 1)
                    qbuf[(((size_t)(bi * NHEAD + h)) * SEQ + t) * HDIM + d] = bv;
                else  // V transposed: [B,H,D,T] so PV B-fragments are contiguous loads
                    vbuf[(((size_t)(bi * NHEAD + h)) * HDIM + d) * SEQ + t] = bv;
            }
        }
    }
}

// ---------------------------------------------------------------------------
// Kernel 2: causal flash attention. One wave per (b, h, 16-query tile).
// Per 32-key block: 4 WMMAs (S = Q*K^T), online softmax, LDS relayout of P,
// 4 WMMAs (O += P*V). fp32 accumulation throughout.
// ---------------------------------------------------------------------------
__global__ __launch_bounds__(32)
void flash_attn_kernel(const __bf16* __restrict__ kbuf,
                       const __bf16* __restrict__ qbuf,
                       const __bf16* __restrict__ vbuf,
                       float* __restrict__ out)
{
    __shared__ __bf16 pLds[16][36];   // padded: 18m mod 64 banks all distinct

    const int lane = threadIdx.x & 31;
    const int half = lane >> 4;
    const int l16  = lane & 15;
    const int qt   = blockIdx.x;              // 0..127
    const int h    = blockIdx.y;
    const int b    = blockIdx.z;
    const int q0   = qt * 16;

    const __bf16* Kh = kbuf + ((size_t)(b * NHEAD + h)) * SEQ * HDIM;  // [t][d]
    const __bf16* Qh = qbuf + ((size_t)(b * NHEAD + h)) * SEQ * HDIM;  // [t][d] (pre-scaled)
    const __bf16* Vh = vbuf + ((size_t)(b * NHEAD + h)) * HDIM * SEQ;  // [d][t]

    // Q fragments: two 16x32 A-frags covering d=0..31, 32..63
    Frag qf[2];
    {
        const __bf16* qrow = Qh + (size_t)(q0 + l16) * HDIM;
#pragma unroll
        for (int c = 0; c < 2; ++c) {
            qf[c].q4[0] = *(const uint4*)(const void*)(qrow + c * 32 + 8 * half);
            qf[c].q4[1] = *(const uint4*)(const void*)(qrow + c * 32 + 16 + 8 * half);
        }
    }

    float mrun[8], lrun[8];
    v8f   o[4];
#pragma unroll
    for (int i = 0; i < 8; ++i) { mrun[i] = -1e30f; lrun[i] = 0.0f; }
#pragma unroll
    for (int nd = 0; nd < 4; ++nd) o[nd] = (v8f){};

    const int nblocks = (q0 + 15) / 32 + 1;
    for (int kb = 0; kb < nblocks; ++kb) {
        const int  kbase = kb * 32;
        const bool last  = (kb == nblocks - 1);

        if (!last) {  // pipeline: prefetch next key block (K and V rows)
            __builtin_prefetch(Kh + (size_t)(kbase + 32 + lane) * HDIM, 0, 0);
            __builtin_prefetch(Vh + (size_t)(lane * 2) * SEQ + kbase + 32, 0, 0);
        }

        // ---- S = Q * K^T : two 16x16 tiles (keys kbase..+15, kbase+16..+31)
        v8f s[2];
#pragma unroll
        for (int t = 0; t < 2; ++t) {
            const __bf16* krow = Kh + (size_t)(kbase + t * 16 + l16) * HDIM;
            v8f st = {};
#pragma unroll
            for (int c = 0; c < 2; ++c) {
                Frag kf;  // B-frag: lane = key column, elems = d (table order)
                kf.q4[0] = *(const uint4*)(const void*)(krow + c * 32 + 8 * half);
                kf.q4[1] = *(const uint4*)(const void*)(krow + c * 32 + 16 + 8 * half);
                st = __builtin_amdgcn_wmma_f32_16x16x32_bf16(
                    false, qf[c].v, false, kf.v, (short)0, st, false, false);
            }
            s[t] = st;
        }

        // ---- causal mask (only the final block can cross the diagonal)
        if (last) {
#pragma unroll
            for (int t = 0; t < 2; ++t) {
                const int kcol = kbase + t * 16 + l16;
#pragma unroll
                for (int i = 0; i < 8; ++i) {
                    const int q = q0 + i + 8 * half;
                    if (kcol > q) s[t][i] = -1e30f;
                }
            }
        }

        // ---- online softmax + stage P (bf16) into LDS in C/D layout
#pragma unroll
        for (int i = 0; i < 8; ++i) {
            float mx   = rowmax16(fmaxf(s[0][i], s[1][i]));
            float mnew = fmaxf(mrun[i], mx);
            float corr = __expf(mrun[i] - mnew);
            float p0   = __expf(s[0][i] - mnew);
            float p1   = __expf(s[1][i] - mnew);
            lrun[i] = lrun[i] * corr + rowsum16(p0 + p1);
            mrun[i] = mnew;
#pragma unroll
            for (int nd = 0; nd < 4; ++nd) o[nd][i] *= corr;
            const int m = i + 8 * half;
            pLds[m][l16]      = tobf(p0);
            pLds[m][16 + l16] = tobf(p1);
        }
        __syncthreads();  // single-wave block: cheap; orders DS store->load

        // ---- reload P as a 16x32 A-fragment
        Frag pf;
#pragma unroll
        for (int j = 0; j < 4; ++j) {
            pf.ui[j]     = *(const unsigned int*)(const void*)&pLds[l16][8 * half + 2 * j];
            pf.ui[4 + j] = *(const unsigned int*)(const void*)&pLds[l16][16 + 8 * half + 2 * j];
        }
        __syncthreads();

        // ---- O += P * V : four 16x16 d-tiles; V is [d][t] so loads are contiguous
#pragma unroll
        for (int nd = 0; nd < 4; ++nd) {
            const __bf16* vrow = Vh + (size_t)(nd * 16 + l16) * SEQ + kbase + 8 * half;
            Frag vf;
            vf.q4[0] = *(const uint4*)(const void*)(vrow);
            vf.q4[1] = *(const uint4*)(const void*)(vrow + 16);
            o[nd] = __builtin_amdgcn_wmma_f32_16x16x32_bf16(
                false, pf.v, false, vf.v, (short)0, o[nd], false, false);
        }
    }

    // ---- normalize and store fp32 output [B,T,C] (coalesced per 16-lane group)
#pragma unroll
    for (int i = 0; i < 8; ++i) {
        const float inv = 1.0f / lrun[i];
        const int   t   = q0 + i + 8 * half;
        float* orow = out + ((size_t)(b * SEQ + t)) * DIMC + h * HDIM;
#pragma unroll
        for (int nd = 0; nd < 4; ++nd)
            orow[nd * 16 + l16] = o[nd][i] * inv;
    }
}

// ---------------------------------------------------------------------------
// Launch
// ---------------------------------------------------------------------------
extern "C" void kernel_launch(void* const* d_in, const int* in_sizes, int n_in,
                              void* d_out, int out_size, void* d_ws, size_t ws_size,
                              hipStream_t stream)
{
    const float* x    = (const float*)d_in[0];   // [B, T, C]
    const float* w    = (const float*)d_in[1];   // [3C, C]
    const float* bias = (const float*)d_in[2];   // [3C]
    float*       out  = (float*)d_out;           // [B, T, C]

    // Workspace layout (bf16): K | Q | V | Xb | Wb   (16+16+16+16+6 = 70 MB)
    const size_t qkv_elems = (size_t)NBATCH * SEQ * DIMC;    // 8M
    const size_t x_elems   = (size_t)NBATCH * SEQ * DIMC;    // 8M
    const size_t w_elems   = (size_t)3 * DIMC * DIMC;        // 3M
    __bf16* kb = (__bf16*)d_ws;
    __bf16* qb = kb + qkv_elems;
    __bf16* vb = qb + qkv_elems;
    __bf16* xb = vb + qkv_elems;
    __bf16* wbf = xb + x_elems;

    // One-shot bf16 conversion of X and W (keeps GEMM hot loop conversion-free)
    cvt_bf16_kernel<<<(int)(x_elems / 4 / 256), 256, 0, stream>>>(x, xb, (int)(x_elems / 4));
    cvt_bf16_kernel<<<(int)(w_elems / 4 / 256), 256, 0, stream>>>(w, wbf, (int)(w_elems / 4));

    // 256 x 96 = 24576 32x32 tiles, 4 waves (tiles) per 128-thread block
    qkv_gemm_kernel<<<24576 / 4, 128, 0, stream>>>(xb, wbf, bias, kb, qb, vb);

    // one 32-thread wave per (b, h, 16-query tile)
    flash_attn_kernel<<<dim3(SEQ / 16, NHEAD, NBATCH), 32, 0, stream>>>(kb, qb, vb, out);
}
